// MoEFused_22995254902975
// MI455X (gfx1250) — matrix operand order repaired and checked by reading, not compiled, not run on hardware
//
#include <hip/hip_runtime.h>
#include <math.h>

// ---------------------------------------------------------------------------
// MoE + shared-expert SwiGLU for MI455X (gfx1250, wave32, WMMA).
// Heavy GEMMs use v_wmma_f32_16x16x32_bf16 (bf16 operands, f32 accumulate).
// A-tiles (bf16 activations) stream into LDS with GLOBAL_LOAD_ASYNC_TO_LDS;
// f32 weights are converted to bf16 in VGPRs and written to LDS transposed
// with packed b64 stores.
// ---------------------------------------------------------------------------
typedef __bf16 bf16_t;
typedef __attribute__((ext_vector_type(8)))  __bf16 v8bf;
typedef __attribute__((ext_vector_type(16))) __bf16 v16bf;
typedef __attribute__((ext_vector_type(8)))  float  v8f;
typedef __attribute__((ext_vector_type(4)))  int    v4i;

#define T_TOK 8192
#define DIM   2048
#define NEXP  8
#define FEXP  1024
#define FSH   1024

#define BM 128
#define BN 128
#define BK 64
#define LDK 72   // padded K stride in LDS (bf16 elems): 144B, 16B aligned

#define AS1 __attribute__((address_space(1)))
#define AS3 __attribute__((address_space(3)))

#if __has_builtin(__builtin_amdgcn_global_load_async_to_lds_b128)
#define USE_ASYNC_COPY 1
#else
#define USE_ASYNC_COPY 0
#endif

union Frag16 { v16bf v; v8bf h[2]; };

static __device__ __forceinline__ void async_wait_zero() {
#if USE_ASYNC_COPY
#if __has_builtin(__builtin_amdgcn_s_wait_asynccnt)
    __builtin_amdgcn_s_wait_asynccnt(0);
#else
    asm volatile("s_wait_asynccnt 0x0" ::: "memory");
#endif
#endif
}

// copy 16B of bf16 activations: global -> LDS (async DMA path if available)
static __device__ __forceinline__ void copy16_g2l(const bf16_t* __restrict__ src,
                                                  bf16_t* __restrict__ dst) {
#if USE_ASYNC_COPY
    __builtin_amdgcn_global_load_async_to_lds_b128(
        (AS1 v4i*)src, (AS3 v4i*)dst, 0, 0);
#else
    *(v8bf*)dst = *(const v8bf*)src;
#endif
}

static __device__ __forceinline__ void zero16_l(bf16_t* __restrict__ dst) {
    union { v8bf v; unsigned int u[4]; } z;
    z.u[0] = z.u[1] = z.u[2] = z.u[3] = 0u;
    *(v8bf*)dst = z.v;
}

static __device__ __forceinline__ unsigned long long pk4_bf16(float a, float b,
                                                              float c, float d) {
    union { __bf16 h[4]; unsigned long long u; } r;
    r.h[0] = (__bf16)a; r.h[1] = (__bf16)b; r.h[2] = (__bf16)c; r.h[3] = (__bf16)d;
    return r.u;
}

// ---------------------------------------------------------------------------
// Kernel 0: zero per-expert counters
// ---------------------------------------------------------------------------
__global__ void init_counts_kernel(int* counts) {
    if (threadIdx.x < NEXP) counts[threadIdx.x] = 0;
}

// ---------------------------------------------------------------------------
// Kernel 1: fused RMSNorm + router (softmax, top-2, renorm) + expert scatter.
// ---------------------------------------------------------------------------
__global__ __launch_bounds__(256)
void rmsnorm_router_kernel(const float* __restrict__ x,
                           const float* __restrict__ rmsw,
                           const float* __restrict__ rw,     // [D, 8]
                           bf16_t* __restrict__ H,           // [T, D] bf16 out
                           int*   __restrict__ counts,       // [8]
                           int*   __restrict__ tlist,        // [8, T]
                           float* __restrict__ twt)          // [8, T]
{
    const int t   = blockIdx.x;
    const int tid = threadIdx.x;
    const float* xr = x + (size_t)t * DIM;

    float ss = 0.f;
    float le[NEXP];
#pragma unroll
    for (int e = 0; e < NEXP; ++e) le[e] = 0.f;

    for (int i = tid; i < DIM; i += 256) {
        float xv = xr[i];
        ss += xv * xv;
        float hv = xv * rmsw[i];
        const float* rwp = rw + (size_t)i * NEXP;
#pragma unroll
        for (int e = 0; e < NEXP; ++e) le[e] += hv * rwp[e];
    }

    __shared__ float sred[NEXP + 1][256];
    sred[0][tid] = ss;
#pragma unroll
    for (int e = 0; e < NEXP; ++e) sred[1 + e][tid] = le[e];
    __syncthreads();
    for (int s = 128; s > 0; s >>= 1) {
        if (tid < s) {
#pragma unroll
            for (int r = 0; r < NEXP + 1; ++r) sred[r][tid] += sred[r][tid + s];
        }
        __syncthreads();
    }

    __shared__ float s_rstd;
    if (tid == 0) s_rstd = rsqrtf(sred[0][0] / (float)DIM + 1e-6f);
    __syncthreads();
    const float rstd = s_rstd;

    for (int i = tid; i < DIM; i += 256)
        H[(size_t)t * DIM + i] = (bf16_t)(xr[i] * rmsw[i] * rstd);

    if (tid == 0) {
        float lg[NEXP], mx = -1e30f;
#pragma unroll
        for (int e = 0; e < NEXP; ++e) { lg[e] = sred[1 + e][0] * rstd; mx = fmaxf(mx, lg[e]); }
        float p[NEXP];
#pragma unroll
        for (int e = 0; e < NEXP; ++e) p[e] = __expf(lg[e] - mx);
        int i0 = 0;
#pragma unroll
        for (int e = 1; e < NEXP; ++e) if (p[e] > p[i0]) i0 = e;
        int i1 = (i0 == 0) ? 1 : 0;
#pragma unroll
        for (int e = 0; e < NEXP; ++e) if (e != i0 && p[e] > p[i1]) i1 = e;
        float inv = 1.f / (p[i0] + p[i1]);
        int p0 = atomicAdd(&counts[i0], 1);
        tlist[i0 * T_TOK + p0] = t;  twt[i0 * T_TOK + p0] = p[i0] * inv;
        int p1 = atomicAdd(&counts[i1], 1);
        tlist[i1 * T_TOK + p1] = t;  twt[i1 * T_TOK + p1] = p[i1] * inv;
    }
}

// ---------------------------------------------------------------------------
// Shared staging helpers for the 128x128x64 GEMM tiles.
//   stage_a : 128 rows x 64 bf16, optional gather by token list, async DMA.
//   stage_b : 64(K) x 128(N) f32 -> bf16, transposed [n][k], b64 LDS stores.
//             Each thread handles a 4(K) x 4(N) micro-tile per pass (2 passes).
// ---------------------------------------------------------------------------
static __device__ __forceinline__ void stage_a_tile(
    bf16_t (*sA)[LDK], const bf16_t* __restrict__ src, int srcStride,
    const int* __restrict__ tl, int m0, int rows, int k0, int tid, bool gather)
{
    const int chunk = tid & 7;      // 8 x 16B per row
    const int rbase = tid >> 3;     // 32 rows / pass
#pragma unroll
    for (int p = 0; p < 4; ++p) {
        int r  = rbase + p * 32;
        int gm = m0 + r;
        bf16_t* dst = &sA[r][chunk * 8];
        if (gm < rows) {
            int row = gather ? tl[gm] : gm;
            copy16_g2l(src + (size_t)row * srcStride + k0 + chunk * 8, dst);
        } else {
            zero16_l(dst);
        }
    }
}

static __device__ __forceinline__ void stage_b_tile(
    bf16_t (*sB)[LDK], const float* __restrict__ W, int ldw,
    int k0, int n0, int tid, bool prefetch_next)
{
    const int nc  = (tid & 31) * 4;  // 4-column group
    const int kq8 = tid >> 5;        // k-quad base 0..7
#pragma unroll
    for (int p = 0; p < 2; ++p) {
        int k = (kq8 + p * 8) * 4;   // k multiple of 4
        const float* base = W + (size_t)(k0 + k) * ldw + n0 + nc;
        float4 r0 = *(const float4*)(base);
        float4 r1 = *(const float4*)(base + ldw);
        float4 r2 = *(const float4*)(base + 2 * (size_t)ldw);
        float4 r3 = *(const float4*)(base + 3 * (size_t)ldw);
        *(unsigned long long*)&sB[nc + 0][k] = pk4_bf16(r0.x, r1.x, r2.x, r3.x);
        *(unsigned long long*)&sB[nc + 1][k] = pk4_bf16(r0.y, r1.y, r2.y, r3.y);
        *(unsigned long long*)&sB[nc + 2][k] = pk4_bf16(r0.z, r1.z, r2.z, r3.z);
        *(unsigned long long*)&sB[nc + 3][k] = pk4_bf16(r0.w, r1.w, r2.w, r3.w);
        if (prefetch_next && p == 0)
            __builtin_prefetch(base + (size_t)BK * ldw, 0, 0);  // global_prefetch_b8
    }
}

// ---------------------------------------------------------------------------
// Kernel 2: gate/up GEMM + SiLU(g)*u  ->  bf16 activations.
// GATHER=true: rows gathered via per-expert token list, grid.z = expert.
// Tile 128x128x64, 8 waves (4x2), wave tile 32x64 (2x4 WMMA 16x16x32 frags).
// ---------------------------------------------------------------------------
template <bool GATHER>
__global__ __launch_bounds__(256)
void gateup_kernel(const bf16_t* __restrict__ H,
                   const float*  __restrict__ WgBase,   // [e, D, F]
                   const float*  __restrict__ WuBase,
                   bf16_t*       __restrict__ actBase,  // [e, T, F]
                   const int*    __restrict__ counts,
                   const int*    __restrict__ tlist,
                   int Fdim)
{
    const int e    = blockIdx.z;
    const int rows = GATHER ? counts[e] : T_TOK;
    const int m0   = blockIdx.x * BM;
    if (m0 >= rows) return;
    const int n0   = blockIdx.y * BN;

    const float* Wg = WgBase + (size_t)e * DIM * Fdim;
    const float* Wu = WuBase + (size_t)e * DIM * Fdim;
    bf16_t*      act = actBase + (size_t)e * T_TOK * Fdim;
    const int*   tl  = tlist + (size_t)e * T_TOK;

    __shared__ bf16_t sA [BM][LDK];
    __shared__ bf16_t sBg[BN][LDK];   // transposed: [n][k]
    __shared__ bf16_t sBu[BN][LDK];

    const int tid  = threadIdx.x;
    const int lane = tid & 31;
    const int wid  = tid >> 5;
    const int wm   = wid >> 1;       // 0..3
    const int wn   = wid & 1;        // 0..1
    const int half = lane >> 4;
    const int l16  = lane & 15;

    v8f cg[2][4], cu[2][4];
#pragma unroll
    for (int i = 0; i < 2; ++i)
#pragma unroll
        for (int j = 0; j < 4; ++j)
#pragma unroll
            for (int v = 0; v < 8; ++v) { cg[i][j][v] = 0.f; cu[i][j][v] = 0.f; }

    for (int k0 = 0; k0 < DIM; k0 += BK) {
        __syncthreads();
        stage_a_tile(sA, H, DIM, tl, m0, rows, k0, tid, GATHER);
        stage_b_tile(sBg, Wg, Fdim, k0, n0, tid, k0 + BK < DIM);
        stage_b_tile(sBu, Wu, Fdim, k0, n0, tid, k0 + BK < DIM);
        async_wait_zero();
        __syncthreads();
#pragma unroll
        for (int kk = 0; kk < BK; kk += 32) {
            Frag16 a[2];
#pragma unroll
            for (int i = 0; i < 2; ++i) {
                const bf16_t* p = &sA[wm * 32 + i * 16 + l16][kk + half * 8];
                a[i].h[0] = *(const v8bf*)p;          // K = half*8 .. +7
                a[i].h[1] = *(const v8bf*)(p + 16);   // K = 16+half*8 .. +7
            }
#pragma unroll
            for (int j = 0; j < 4; ++j) {
                const bf16_t* pg = &sBg[wn * 64 + j * 16 + l16][kk + half * 8];
                Frag16 bg; bg.h[0] = *(const v8bf*)pg; bg.h[1] = *(const v8bf*)(pg + 16);
                const bf16_t* pu = &sBu[wn * 64 + j * 16 + l16][kk + half * 8];
                Frag16 bu; bu.h[0] = *(const v8bf*)pu; bu.h[1] = *(const v8bf*)(pu + 16);
#pragma unroll
                for (int i = 0; i < 2; ++i) {
                    cg[i][j] = __builtin_amdgcn_wmma_f32_16x16x32_bf16(
                        false, a[i].v, false, bg.v, (short)0, cg[i][j], false, false);
                    cu[i][j] = __builtin_amdgcn_wmma_f32_16x16x32_bf16(
                        false, a[i].v, false, bu.v, (short)0, cu[i][j], false, false);
                }
            }
        }
    }

    // epilogue: act = silu(g) * u, bf16, compact rows
#pragma unroll
    for (int i = 0; i < 2; ++i)
#pragma unroll
        for (int j = 0; j < 4; ++j)
#pragma unroll
            for (int v = 0; v < 8; ++v) {
                int ml = wm * 32 + i * 16 + half * 8 + v;
                int nl = wn * 64 + j * 16 + l16;
                int gm = m0 + ml;
                if (gm < rows) {
                    float g = cg[i][j][v], u = cu[i][j][v];
                    float s = g / (1.f + __expf(-g));
                    act[(size_t)gm * Fdim + n0 + nl] = (bf16_t)(s * u);
                }
            }
}

// ---------------------------------------------------------------------------
// Kernel 3: down GEMM.
// ATOMIC=false: shared expert, plain stores (initializes d_out everywhere).
// ATOMIC=true : per-expert (grid.z), scaled by router weight, scattered back
//               to out[token] via global_atomic_add_f32.
// ---------------------------------------------------------------------------
template <bool ATOMIC>
__global__ __launch_bounds__(256)
void down_kernel(const bf16_t* __restrict__ actBase,  // [e, T, Kd] compact
                 const float*  __restrict__ WdBase,   // [e, Kd, D]
                 float*        __restrict__ out,      // [T, D]
                 const int*    __restrict__ counts,
                 const int*    __restrict__ tlist,
                 const float*  __restrict__ twt,
                 int Kd)
{
    const int e    = blockIdx.z;
    const int rows = ATOMIC ? counts[e] : T_TOK;
    const int m0   = blockIdx.x * BM;
    if (m0 >= rows) return;
    const int n0   = blockIdx.y * BN;

    const bf16_t* A  = actBase + (size_t)e * T_TOK * Kd;
    const float*  Wd = WdBase + (size_t)e * Kd * DIM;

    __shared__ bf16_t sA[BM][LDK];
    __shared__ bf16_t sB[BN][LDK];   // transposed [n][k]

    const int tid  = threadIdx.x;
    const int lane = tid & 31;
    const int wid  = tid >> 5;
    const int wm   = wid >> 1;
    const int wn   = wid & 1;
    const int half = lane >> 4;
    const int l16  = lane & 15;

    v8f c[2][4];
#pragma unroll
    for (int i = 0; i < 2; ++i)
#pragma unroll
        for (int j = 0; j < 4; ++j)
#pragma unroll
            for (int v = 0; v < 8; ++v) c[i][j][v] = 0.f;

    for (int k0 = 0; k0 < Kd; k0 += BK) {
        __syncthreads();
        stage_a_tile(sA, A, Kd, tlist /*unused*/, m0, rows, k0, tid, false);
        stage_b_tile(sB, Wd, DIM, k0, n0, tid, k0 + BK < Kd);
        async_wait_zero();
        __syncthreads();
#pragma unroll
        for (int kk = 0; kk < BK; kk += 32) {
            Frag16 a[2];
#pragma unroll
            for (int i = 0; i < 2; ++i) {
                const bf16_t* p = &sA[wm * 32 + i * 16 + l16][kk + half * 8];
                a[i].h[0] = *(const v8bf*)p;
                a[i].h[1] = *(const v8bf*)(p + 16);
            }
#pragma unroll
            for (int j = 0; j < 4; ++j) {
                const bf16_t* pb = &sB[wn * 64 + j * 16 + l16][kk + half * 8];
                Frag16 b; b.h[0] = *(const v8bf*)pb; b.h[1] = *(const v8bf*)(pb + 16);
#pragma unroll
                for (int i = 0; i < 2; ++i)
                    c[i][j] = __builtin_amdgcn_wmma_f32_16x16x32_bf16(
                        false, a[i].v, false, b.v, (short)0, c[i][j], false, false);
            }
        }
    }

#pragma unroll
    for (int i = 0; i < 2; ++i)
#pragma unroll
        for (int j = 0; j < 4; ++j)
#pragma unroll
            for (int v = 0; v < 8; ++v) {
                int ml = wm * 32 + i * 16 + half * 8 + v;
                int nl = wn * 64 + j * 16 + l16;
                int gm = m0 + ml;
                if (gm < rows) {
                    if (ATOMIC) {
                        int   token = tlist[(size_t)e * T_TOK + gm];
                        float w     = twt[(size_t)e * T_TOK + gm];
                        atomicAdd(&out[(size_t)token * DIM + n0 + nl], c[i][j][v] * w);
                    } else {
                        out[(size_t)gm * DIM + n0 + nl] = c[i][j][v];
                    }
                }
            }
}

// ---------------------------------------------------------------------------
// Host-side launch. Workspace layout (~185 MB):
//   H : T*D bf16 | actS : T*FSH bf16 | actE : E*T*FEXP bf16 (worst case)
//   counts : E ints | tlist/twt : E*T each
// ---------------------------------------------------------------------------
extern "C" void kernel_launch(void* const* d_in, const int* in_sizes, int n_in,
                              void* d_out, int out_size, void* d_ws, size_t ws_size,
                              hipStream_t stream) {
    const float* x       = (const float*)d_in[0];
    const float* rms_w   = (const float*)d_in[1];
    const float* rw      = (const float*)d_in[2];
    const float* w_gate  = (const float*)d_in[3];
    const float* w_up    = (const float*)d_in[4];
    const float* w_down  = (const float*)d_in[5];
    const float* sw_gate = (const float*)d_in[6];
    const float* sw_up   = (const float*)d_in[7];
    const float* sw_down = (const float*)d_in[8];
    float* out = (float*)d_out;

    char* ws = (char*)d_ws;
    size_t off = 0;
    bf16_t* H    = (bf16_t*)(ws + off); off += (size_t)T_TOK * DIM * sizeof(bf16_t);
    bf16_t* actS = (bf16_t*)(ws + off); off += (size_t)T_TOK * FSH * sizeof(bf16_t);
    bf16_t* actE = (bf16_t*)(ws + off); off += (size_t)NEXP * T_TOK * FEXP * sizeof(bf16_t);
    int*    counts = (int*)(ws + off);  off += 256;
    int*    tlist  = (int*)(ws + off);  off += (size_t)NEXP * T_TOK * sizeof(int);
    float*  twt    = (float*)(ws + off);

    init_counts_kernel<<<1, 32, 0, stream>>>(counts);

    rmsnorm_router_kernel<<<T_TOK, 256, 0, stream>>>(
        x, rms_w, rw, H, counts, tlist, twt);

    // shared expert: gate/up + silu*u, then down (plain stores initialize out)
    gateup_kernel<false><<<dim3(T_TOK / BM, FSH / BN, 1), 256, 0, stream>>>(
        H, sw_gate, sw_up, actS, nullptr, tlist, FSH);
    down_kernel<false><<<dim3(T_TOK / BM, DIM / BN, 1), 256, 0, stream>>>(
        actS, sw_down, out, nullptr, tlist, twt, FSH);

    // routed experts: grid.z = expert; tiles past counts[e] exit immediately
    gateup_kernel<true><<<dim3(T_TOK / BM, FEXP / BN, NEXP), 256, 0, stream>>>(
        H, w_gate, w_up, actE, counts, tlist, FEXP);
    down_kernel<true><<<dim3(T_TOK / BM, DIM / BN, NEXP), 256, 0, stream>>>(
        actE, w_down, out, counts, tlist, twt, FEXP);
}
